// Group_MLP_43903155700155
// MI455X (gfx1250) — matrix operand order repaired and checked by reading, not compiled, not run on hardware
//
#include <hip/hip_runtime.h>

#define ROWS    64
#define IN_DIM  1024
#define OUT_DIM 1024
#define NBATCH  512

#define BM 128   // batch-tile (M)
#define BN 128   // out-tile   (N)
#define KT 32    // k-tile (bf16 WMMA K)
#define LDSTR 20 // dwords per LDS row: 16 data + 4 pad

// dynamic-LDS layout (dwords)
#define ARRDW  (BM * LDSTR)   // 2560 dwords per array
#define XL_OFF (ARRDW)        // 2560
#define WH_OFF (2 * ARRDW)    // 5120
#define WL_OFF (3 * ARRDW)    // 7680
#define BUFDW  (4 * ARRDW)    // 10240 dwords per buffer (40 KB)

typedef __attribute__((ext_vector_type(16))) __bf16 v16bf;
typedef __attribute__((ext_vector_type(8)))  float  v8f;

union FragAB {
    uint4 q[2];
    v16bf v;
};

// fp32 -> bf16 round-to-nearest-even (16-bit pattern in low bits)
__device__ __forceinline__ unsigned int bf16_rne(float f) {
    unsigned int u = __builtin_bit_cast(unsigned int, f);
    return (u + 0x7FFFu + ((u >> 16) & 1u)) >> 16;
}
__device__ __forceinline__ float bf16_up(unsigned int h) {
    return __builtin_bit_cast(float, h << 16);
}
// split two fp32 into packed-pair bf16 hi and lo words (a in low half)
__device__ __forceinline__ void split2(float a, float b, unsigned int& hi, unsigned int& lo) {
    unsigned int ha = bf16_rne(a), hb = bf16_rne(b);
    float ra = a - bf16_up(ha);
    float rb = b - bf16_up(hb);
    unsigned int la = bf16_rne(ra), lb = bf16_rne(rb);
    hi = ha | (hb << 16);
    lo = la | (lb << 16);
}

// ============================================================================
// Prep kernel 1: x (fp32, [b][r][k]) -> packed bf16 hi/lo, same layout
// ============================================================================
__global__ __launch_bounds__(256)
void split_x_kernel(const float* __restrict__ X,
                    unsigned int* __restrict__ XH,
                    unsigned int* __restrict__ XL)
{
    size_t i4 = (size_t)blockIdx.x * 256 + threadIdx.x;   // float4 index
    float4 v = *((const float4*)X + i4);
    unsigned int h01, l01, h23, l23;
    split2(v.x, v.y, h01, l01);
    split2(v.z, v.w, h23, l23);
    size_t d = i4 * 2;
    XH[d] = h01;  XH[d + 1] = h23;
    XL[d] = l01;  XL[d + 1] = l23;
}

// ============================================================================
// Prep kernel 2: W (fp32, [r][k][n]) -> bf16 hi/lo TRANSPOSED to [r][n][k]
// ============================================================================
__global__ __launch_bounds__(256)
void split_w_kernel(const float* __restrict__ W,
                    unsigned int* __restrict__ WTH,
                    unsigned int* __restrict__ WTL)
{
    __shared__ float T[32][33];

    const int n0 = blockIdx.x * 32;
    const int k0 = blockIdx.y * 32;
    const int r  = blockIdx.z;
    const int t  = threadIdx.x;

    {
        int k = t >> 3;
        int nc = (t & 7) * 4;
        float4 v = *(const float4*)(W + ((size_t)r * IN_DIM + k0 + k) * OUT_DIM + n0 + nc);
        T[k][nc]     = v.x;
        T[k][nc + 1] = v.y;
        T[k][nc + 2] = v.z;
        T[k][nc + 3] = v.w;
    }
    __syncthreads();
    {
        int n  = t >> 3;
        int kc = (t & 7) * 4;
        float f0 = T[kc][n], f1 = T[kc + 1][n], f2 = T[kc + 2][n], f3 = T[kc + 3][n];
        unsigned int h01, l01, h23, l23;
        split2(f0, f1, h01, l01);
        split2(f2, f3, h23, l23);
        size_t d = (((size_t)r * OUT_DIM + n0 + n) * IN_DIM + k0 + kc) >> 1;
        WTH[d] = h01;  WTH[d + 1] = h23;
        WTL[d] = l01;  WTL[d + 1] = l23;
    }
}

// ============================================================================
// Main GEMM: pre-split bf16 operands, double-buffered dynamic LDS (80 KB).
// Staging is a compact global->LDS copy at the top of each iteration (staged
// registers die at the ds_store => no cross-compute liveness => no spills).
// The blocking loads are L2/WGP$ hits thanks to global_prefetch_b8 issued one
// iteration ahead (distance k+2).
// ============================================================================
__global__ __launch_bounds__(256)
void Group_MLP_wmma_kernel(const char* __restrict__ XHg,
                           const char* __restrict__ XLg,
                           const char* __restrict__ WHg,
                           const char* __restrict__ WLg,
                           const float* __restrict__ Bv,
                           float* __restrict__ O)
{
    extern __shared__ __align__(16) unsigned int lds[];   // 2 * BUFDW dwords

    const int n0 = blockIdx.x * BN;
    const int b0 = blockIdx.y * BM;
    const int r  = blockIdx.z;

    const int t    = threadIdx.x;
    const int wave = t >> 5;
    const int lane = t & 31;
    const int wm   = wave >> 1;   // 0..3 : 32-row slab
    const int wn   = wave & 1;    // 0..1 : 64-col slab
    const int l16  = lane & 15;
    const int half = lane >> 4;

    // staging map: row = t>>1 (128 rows), 32B half-row = (t&1)*32 bytes
    const int srow  = t >> 1;
    const int shalf = (t & 1) * 32;
    const int sldsb = srow * LDSTR + (t & 1) * 8;

    const size_t xrow = (((size_t)(b0 + srow) * ROWS + r) * IN_DIM) * 2 + shalf;
    const size_t wrow = (((size_t)r * OUT_DIM + n0 + srow) * IN_DIM) * 2 + shalf;

    // direct global->LDS copy of tile ks into buffer p (registers die here)
    auto stage = [&](int ks, int p) {
        size_t xo = xrow + (size_t)ks * (KT * 2);
        size_t wo = wrow + (size_t)ks * (KT * 2);
        unsigned int* B = lds + p * BUFDW;
        uint4 a0 = *(const uint4*)(XHg + xo);
        uint4 a1 = *(const uint4*)(XHg + xo + 16);
        uint4 a2 = *(const uint4*)(XLg + xo);
        uint4 a3 = *(const uint4*)(XLg + xo + 16);
        uint4 a4 = *(const uint4*)(WHg + wo);
        uint4 a5 = *(const uint4*)(WHg + wo + 16);
        uint4 a6 = *(const uint4*)(WLg + wo);
        uint4 a7 = *(const uint4*)(WLg + wo + 16);
        *(uint4*)&B[sldsb]              = a0;
        *(uint4*)&B[sldsb + 4]          = a1;
        *(uint4*)&B[XL_OFF + sldsb]     = a2;
        *(uint4*)&B[XL_OFF + sldsb + 4] = a3;
        *(uint4*)&B[WH_OFF + sldsb]     = a4;
        *(uint4*)&B[WH_OFF + sldsb + 4] = a5;
        *(uint4*)&B[WL_OFF + sldsb]     = a6;
        *(uint4*)&B[WL_OFF + sldsb + 4] = a7;
    };

    v8f acc[2][4];
#pragma unroll
    for (int mt = 0; mt < 2; ++mt)
#pragma unroll
        for (int nt = 0; nt < 4; ++nt)
            acc[mt][nt] = (v8f){0.f, 0.f, 0.f, 0.f, 0.f, 0.f, 0.f, 0.f};

    const int NK = IN_DIM / KT;

    stage(0, 0);
    __syncthreads();

#pragma unroll 1
    for (int ks = 0; ks < NK; ++ks) {
        const int cur = (ks & 1) * BUFDW;

        // copy next tile into the other buffer (loads are cache hits via the
        // prefetch issued last iteration); registers die at the ds_store
        if (ks + 1 < NK) stage(ks + 1, 1 - (ks & 1));

        // prefetch tile k+2 into WGP$/L2 (global_prefetch_b8)
        if (ks + 2 < NK) {
            size_t xo = xrow + (size_t)(ks + 2) * (KT * 2);
            size_t wo = wrow + (size_t)(ks + 2) * (KT * 2);
            __builtin_prefetch(XHg + xo, 0, 1);
            __builtin_prefetch(XLg + xo, 0, 1);
            __builtin_prefetch(WHg + wo, 0, 1);
            __builtin_prefetch(WLg + wo, 0, 1);
        }

        // A fragments (hi & lo) for this wave's two 16-row slabs
        FragAB ah[2], al[2];
#pragma unroll
        for (int mt = 0; mt < 2; ++mt) {
            int row  = wm * 32 + mt * 16 + l16;
            int base = cur + row * LDSTR + half * 4;
            ah[mt].q[0] = *(const uint4*)&lds[base];
            ah[mt].q[1] = *(const uint4*)&lds[base + 8];
            al[mt].q[0] = *(const uint4*)&lds[XL_OFF + base];
            al[mt].q[1] = *(const uint4*)&lds[XL_OFF + base + 8];
        }

        // B fragments just-in-time per nt (caps register pressure)
#pragma unroll
        for (int nt = 0; nt < 4; ++nt) {
            FragAB bh, bl;
            int col  = wn * 64 + nt * 16 + l16;
            int base = cur + col * LDSTR + half * 4;
            bh.q[0] = *(const uint4*)&lds[WH_OFF + base];
            bh.q[1] = *(const uint4*)&lds[WH_OFF + base + 8];
            bl.q[0] = *(const uint4*)&lds[WL_OFF + base];
            bl.q[1] = *(const uint4*)&lds[WL_OFF + base + 8];
#pragma unroll
            for (int mt = 0; mt < 2; ++mt) {
                acc[mt][nt] = __builtin_amdgcn_wmma_f32_16x16x32_bf16(
                    false, ah[mt].v, false, bh.v, (short)0, acc[mt][nt], false, false);
                acc[mt][nt] = __builtin_amdgcn_wmma_f32_16x16x32_bf16(
                    false, al[mt].v, false, bh.v, (short)0, acc[mt][nt], false, false);
                acc[mt][nt] = __builtin_amdgcn_wmma_f32_16x16x32_bf16(
                    false, ah[mt].v, false, bl.v, (short)0, acc[mt][nt], false, false);
            }
        }

        __syncthreads();   // next-buffer stores visible; reads of cur complete
    }

    // epilogue: bias + store (C layout: N = lane%16, M = j + 8*half)
#pragma unroll
    for (int mt = 0; mt < 2; ++mt) {
#pragma unroll
        for (int nt = 0; nt < 4; ++nt) {
            int n = n0 + wn * 64 + nt * 16 + l16;
            float bias = Bv[(size_t)r * OUT_DIM + n];
            int m = b0 + wm * 32 + mt * 16 + half * 8;
            v8f a = acc[mt][nt];
#pragma unroll
            for (int j = 0; j < 8; ++j)
                O[((size_t)(m + j) * ROWS + r) * OUT_DIM + n] = a[j] + bias;
        }
    }
}

// ============================================================================
// Fallback: fused-conversion kernel (used when workspace is too small)
// ============================================================================
__global__ __launch_bounds__(256)
void Group_MLP_fused_kernel(const float* __restrict__ X,
                            const float* __restrict__ W,
                            const float* __restrict__ Bv,
                            float* __restrict__ O)
{
    __shared__ __align__(16) unsigned int XH[BM * LDSTR];
    __shared__ __align__(16) unsigned int XL[BM * LDSTR];
    __shared__ __align__(16) unsigned int WH[BN * LDSTR];
    __shared__ __align__(16) unsigned int WL[BN * LDSTR];

    const int n0 = blockIdx.x * BN;
    const int b0 = blockIdx.y * BM;
    const int r  = blockIdx.z;

    const int t    = threadIdx.x;
    const int wave = t >> 5;
    const int lane = t & 31;
    const int wm   = wave >> 1;
    const int wn   = wave & 1;
    const int l16  = lane & 15;
    const int half = lane >> 4;

    const int xm = t >> 3;
    const int xk = (t & 7) * 4;
    const int wncol = t & 127;
    const int wkz   = (t >> 7) * 8;

    const float* xbase = X + ((size_t)b0 * ROWS + r) * IN_DIM;
    const float* wbase = W + (size_t)r * IN_DIM * OUT_DIM + n0;

    // fused load+convert+store, no registers live across compute
    auto stage = [&](int ks) {
        const float* xp = xbase + ks * KT + xk;
#pragma unroll
        for (int p = 0; p < 4; ++p) {
            float4 v = *(const float4*)(xp + (size_t)(xm + p * 32) * (ROWS * IN_DIM));
            unsigned int h01, l01, h23, l23;
            split2(v.x, v.y, h01, l01);
            split2(v.z, v.w, h23, l23);
            int off = (xm + p * 32) * LDSTR + (xk >> 1);
            XH[off]     = h01;  XH[off + 1] = h23;
            XL[off]     = l01;  XL[off + 1] = l23;
        }
        const float* wp = wbase + (size_t)(ks * KT) * OUT_DIM + wncol;
#pragma unroll
        for (int j = 0; j < 8; ++j) {
            int k2 = (wkz + j) * 2;
            float w0 = wp[(size_t)k2 * OUT_DIM];
            float w1 = wp[(size_t)(k2 + 1) * OUT_DIM];
            unsigned int h, l;
            split2(w0, w1, h, l);
            int off = wncol * LDSTR + wkz + j;
            WH[off] = h;
            WL[off] = l;
        }
    };

    v8f acc[2][4];
#pragma unroll
    for (int mt = 0; mt < 2; ++mt)
#pragma unroll
        for (int nt = 0; nt < 4; ++nt)
            acc[mt][nt] = (v8f){0.f, 0.f, 0.f, 0.f, 0.f, 0.f, 0.f, 0.f};

    const int NK = IN_DIM / KT;

#pragma unroll 1
    for (int ks = 0; ks < NK; ++ks) {
        stage(ks);
        __syncthreads();

        FragAB ah[2], al[2];
#pragma unroll
        for (int mt = 0; mt < 2; ++mt) {
            int row  = wm * 32 + mt * 16 + l16;
            int base = row * LDSTR + half * 4;
            ah[mt].q[0] = *(const uint4*)&XH[base];
            ah[mt].q[1] = *(const uint4*)&XH[base + 8];
            al[mt].q[0] = *(const uint4*)&XL[base];
            al[mt].q[1] = *(const uint4*)&XL[base + 8];
        }
#pragma unroll
        for (int nt = 0; nt < 4; ++nt) {
            FragAB bh, bl;
            int col  = wn * 64 + nt * 16 + l16;
            int base = col * LDSTR + half * 4;
            bh.q[0] = *(const uint4*)&WH[base];
            bh.q[1] = *(const uint4*)&WH[base + 8];
            bl.q[0] = *(const uint4*)&WL[base];
            bl.q[1] = *(const uint4*)&WL[base + 8];
#pragma unroll
            for (int mt = 0; mt < 2; ++mt) {
                acc[mt][nt] = __builtin_amdgcn_wmma_f32_16x16x32_bf16(
                    false, ah[mt].v, false, bh.v, (short)0, acc[mt][nt], false, false);
                acc[mt][nt] = __builtin_amdgcn_wmma_f32_16x16x32_bf16(
                    false, al[mt].v, false, bh.v, (short)0, acc[mt][nt], false, false);
                acc[mt][nt] = __builtin_amdgcn_wmma_f32_16x16x32_bf16(
                    false, ah[mt].v, false, bl.v, (short)0, acc[mt][nt], false, false);
            }
        }

        __syncthreads();
    }

#pragma unroll
    for (int mt = 0; mt < 2; ++mt) {
#pragma unroll
        for (int nt = 0; nt < 4; ++nt) {
            int n = n0 + wn * 64 + nt * 16 + l16;
            float bias = Bv[(size_t)r * OUT_DIM + n];
            int m = b0 + wm * 32 + mt * 16 + half * 8;
            v8f a = acc[mt][nt];
#pragma unroll
            for (int j = 0; j < 8; ++j)
                O[((size_t)(m + j) * ROWS + r) * OUT_DIM + n] = a[j] + bias;
        }
    }
}

extern "C" void kernel_launch(void* const* d_in, const int* in_sizes, int n_in,
                              void* d_out, int out_size, void* d_ws, size_t ws_size,
                              hipStream_t stream) {
    const float* x = (const float*)d_in[0];
    const float* W = (const float*)d_in[1];
    const float* b = (const float*)d_in[2];
    float* out = (float*)d_out;

    const size_t xBytes = (size_t)NBATCH * ROWS * IN_DIM * 2;       // 64 MiB
    const size_t wBytes = (size_t)ROWS * IN_DIM * OUT_DIM * 2;      // 128 MiB
    const size_t need = 2 * xBytes + 2 * wBytes;                    // 384 MiB

    dim3 grid(OUT_DIM / BN, NBATCH / BM, ROWS);
    dim3 block(256);

    if (ws_size >= need) {
        char* ws  = (char*)d_ws;
        char* xh  = ws;
        char* xl  = ws + xBytes;
        char* wth = ws + 2 * xBytes;
        char* wtl = ws + 2 * xBytes + wBytes;

        size_t nf4 = (size_t)NBATCH * ROWS * IN_DIM / 4;
        split_x_kernel<<<dim3((unsigned)(nf4 / 256)), block, 0, stream>>>(
            x, (unsigned int*)xh, (unsigned int*)xl);

        split_w_kernel<<<dim3(OUT_DIM / 32, IN_DIM / 32, ROWS), block, 0, stream>>>(
            W, (unsigned int*)wth, (unsigned int*)wtl);

        const int ldsBytes = 2 * BUFDW * 4;   // 80 KB double buffer
        (void)hipFuncSetAttribute((const void*)Group_MLP_wmma_kernel,
                                  hipFuncAttributeMaxDynamicSharedMemorySize, ldsBytes);
        Group_MLP_wmma_kernel<<<grid, block, ldsBytes, stream>>>(
            xh, xl, wth, wtl, b, out);
    } else {
        Group_MLP_fused_kernel<<<grid, block, 0, stream>>>(x, W, b, out);
    }
}
